// AdaptiveAttentionTransformerBlock_62843961475767
// MI455X (gfx1250) — matrix operand order repaired and checked by reading, not compiled, not run on hardware
//
#include <hip/hip_runtime.h>

// Problem constants (from reference setup_inputs)
#define BATCH   2
#define S_LEN   2048
#define EMB     1024
#define HEADS   16
#define HDIM    64
#define STRIDE  64
#define CHUNKS  32          // S_LEN / STRIDE

typedef __attribute__((ext_vector_type(16))) _Float16 v16h;
typedef __attribute__((ext_vector_type(8)))  float    v8f;

// ---------------------------------------------------------------------------
// f32 -> f16 elementwise convert (4 elems / thread)
// ---------------------------------------------------------------------------
__global__ __launch_bounds__(256) void cvt_f32_to_f16(const float* __restrict__ s,
                                                      _Float16* __restrict__ d, int n) {
  int i = (blockIdx.x * 256 + threadIdx.x) * 4;
  if (i >= n) return;
  float4 f = *(const float4*)(s + i);
  union { _Float16 h[4]; uint2 u; } t;
  t.h[0] = (_Float16)f.x; t.h[1] = (_Float16)f.y;
  t.h[2] = (_Float16)f.z; t.h[3] = (_Float16)f.w;
  *(uint2*)(d + i) = t.u;
}

// ---------------------------------------------------------------------------
// WMMA GEMM, f16 operands:  Out[M,N](f32) = A[M,K] @ W[N,K]^T
// Block: 256 threads (8 waves), 128x128 tile, K step 32, double-buffered LDS
// filled with GLOBAL_LOAD_ASYNC_TO_LDS_B128 (ASYNCcnt pipelined).
// Wave w: rows 32*(w>>1)..+31, cols 64*(w&1)..+63 -> 2x4 WMMA accumulators.
// ---------------------------------------------------------------------------
#define LDT 48   // padded LDS row stride in f16 (96B, 16B-aligned chunks)

__device__ __forceinline__ void issue_tile_async(const _Float16* __restrict__ gbase,
                                                 unsigned ldsBase, int K,
                                                 int blockRow, int kk, int tid) {
#pragma unroll
  for (int q = 0; q < 2; ++q) {
    int cid = tid + q * 256;            // 0..511 : 16B chunk id of 128x32 tile
    int row = cid >> 2;                 // 0..127
    int c8  = (cid & 3) * 8;            // f16 col offset 0,8,16,24
    unsigned lo = ldsBase + (unsigned)(row * LDT + c8) * 2u;
    unsigned long long ga =
        (unsigned long long)(uintptr_t)(gbase + (size_t)(blockRow + row) * K + kk + c8);
    asm volatile("global_load_async_to_lds_b128 %0, %1, off"
                 :: "v"(lo), "v"(ga) : "memory");
  }
}

__global__ __launch_bounds__(256) void gemm_f16_wmma(
    const _Float16* __restrict__ A, const _Float16* __restrict__ W,
    float* __restrict__ Out, int M, int N, int K) {
  __shared__ _Float16 aT[2][128 * LDT];
  __shared__ _Float16 wT[2][128 * LDT];

  const int tid    = threadIdx.x;
  const int wave   = tid >> 5;
  const int lane   = tid & 31;
  const int blockM = blockIdx.y * 128;
  const int blockN = blockIdx.x * 128;
  const int m0     = (wave >> 1) * 32;   // wave row base (2 M-tiles)
  const int n0     = (wave & 1) * 64;    // wave col base (4 N-tiles)

  v8f acc[2][4];
#pragma unroll
  for (int mi = 0; mi < 2; ++mi)
#pragma unroll
    for (int ni = 0; ni < 4; ++ni) acc[mi][ni] = (v8f){};

  const unsigned ldsA0 = (unsigned)(uintptr_t)&aT[0][0];
  const unsigned ldsA1 = (unsigned)(uintptr_t)&aT[1][0];
  const unsigned ldsW0 = (unsigned)(uintptr_t)&wT[0][0];
  const unsigned ldsW1 = (unsigned)(uintptr_t)&wT[1][0];

  const int nsteps = K >> 5;

  // prologue: prefetch first K-slice into buffer 0
  issue_tile_async(A, ldsA0, K, blockM, 0, tid);
  issue_tile_async(W, ldsW0, K, blockN, 0, tid);

  int parity = 0;
  for (int step = 0; step < nsteps; ++step) {
    if (step + 1 < nsteps) {
      int kkn = (step + 1) << 5;
      issue_tile_async(A, parity ? ldsA0 : ldsA1, K, blockM, kkn, tid);
      issue_tile_async(W, parity ? ldsW0 : ldsW1, K, blockN, kkn, tid);
      asm volatile("s_wait_asynccnt 0x4" ::: "memory");
    } else {
      asm volatile("s_wait_asynccnt 0x0" ::: "memory");
    }
    __syncthreads();   // all waves' async writes to buf[parity] now visible

    const _Float16* at = &aT[parity][0];
    const _Float16* wt = &wT[parity][0];

    union Frag { v16h v; unsigned u[8]; };
    Frag a[2], b[4];
    const int lrow = lane & 15;
    const int hi8  = (lane >> 4) * 8;
    const int khi  = (lane >> 4) * 16;
#pragma unroll
    for (int mi = 0; mi < 2; ++mi) {
      const int am = m0 + mi * 16 + lrow;
#pragma unroll
      for (int j = 0; j < 8; ++j) {
        int ka = ((j >= 4) ? 16 : 0) + hi8 + 2 * (j & 3);   // A 16x32 f16 layout
        a[mi].u[j] = *(const unsigned*)&at[am * LDT + ka];
      }
    }
#pragma unroll
    for (int ni = 0; ni < 4; ++ni) {
      const int bn = n0 + ni * 16 + lrow;
#pragma unroll
      for (int j = 0; j < 8; ++j) {
        int kb = ((j >= 4) ? 16 : 0) + hi8 + 2 * (j & 3);   // B 32x16 f16 layout
        b[ni].u[j] = *(const unsigned*)&wt[bn * LDT + kb];
      }
    }
#pragma unroll
    for (int mi = 0; mi < 2; ++mi)
#pragma unroll
      for (int ni = 0; ni < 4; ++ni)
        acc[mi][ni] = __builtin_amdgcn_wmma_f32_16x16x32_f16(
            false, a[mi].v, false, b[ni].v, (short)0, acc[mi][ni], false, false);

    __syncthreads();   // reads done before buf[parity] is refilled
    parity ^= 1;
  }

  // C/D layout: lane&15 = N, VGPR j -> M = j + 8*(lane>=16)
  const int outN = lane & 15;
  const int mAdd = (lane >> 4) * 8;
#pragma unroll
  for (int mi = 0; mi < 2; ++mi)
#pragma unroll
    for (int ni = 0; ni < 4; ++ni) {
#pragma unroll
      for (int j = 0; j < 8; ++j) {
        int m = blockM + m0 + mi * 16 + mAdd + j;
        Out[(size_t)m * N + blockN + n0 + ni * 16 + outN] = acc[mi][ni][j];
      }
    }
}

// ---------------------------------------------------------------------------
// In-place interleaved RoPE on q and k inside qkv[B,S,3,H,D] (f32).
// ---------------------------------------------------------------------------
__global__ __launch_bounds__(256) void rope_kernel(float* __restrict__ qkv) {
  int id = blockIdx.x * 256 + threadIdx.x;   // B*S*2*H*32 threads
  int i     = id & 31;
  int h     = (id >> 5) & 15;
  int which = (id >> 9) & 1;                 // 0 = q, 1 = k
  int s     = (id >> 10) & (S_LEN - 1);
  int b     = id >> 21;
  size_t base = (size_t)(b * S_LEN + s) * (3 * EMB) + which * EMB + h * HDIM;
  int d0 = 2 * i, d1 = 2 * i + 1;
  const float C = 0.28782313662425572f;      // ln(10000)/32
  float a0 = (float)s * expf(-(float)(d0 & 31) * C);
  float a1 = (float)s * expf(-(float)(d1 & 31) * C);
  float x0 = qkv[base + d0], x1 = qkv[base + d1];
  qkv[base + d0] = x0 * cosf(a0) - x1 * sinf(a0);
  qkv[base + d1] = x1 * cosf(a1) + x0 * sinf(a1);
}

// ---------------------------------------------------------------------------
// Chunk-mean latents
// ---------------------------------------------------------------------------
__global__ __launch_bounds__(256) void lat_kernel(const float* __restrict__ qkv,
                                                  float* __restrict__ klat,
                                                  float* __restrict__ vlat) {
  int id = blockIdx.x * 256 + threadIdx.x;   // 2 * B*H*CHUNKS*HDIM threads
  int which = id >> 16;                      // 0 = k, 1 = v
  int d = id & 63;
  int c = (id >> 6) & 31;
  int h = (id >> 11) & 15;
  int b = (id >> 15) & 1;
  size_t src = ((size_t)(b * S_LEN + c * STRIDE) * 3 + 1 + which) * EMB + h * HDIM + d;
  float sum = 0.f;
  for (int j = 0; j < STRIDE; ++j) sum += qkv[src + (size_t)j * 3 * EMB];
  float* dst = which ? vlat : klat;
  dst[(((size_t)b * HEADS + h) * CHUNKS + c) * HDIM + d] = sum * (1.0f / STRIDE);
}

__device__ __forceinline__ void store64_f16(_Float16* __restrict__ dst, const float* o) {
#pragma unroll
  for (int g = 0; g < 8; ++g) {
    union { _Float16 h[8]; uint4 u; } t;
#pragma unroll
    for (int e = 0; e < 8; ++e) t.h[e] = (_Float16)o[g * 8 + e];
    *(uint4*)(dst + g * 8) = t.u;
  }
}

// ---------------------------------------------------------------------------
// Compressed attention -> cath[b,s, h*128 + 0..63] (f16)
// ---------------------------------------------------------------------------
__global__ __launch_bounds__(256) void comp_attn(const float* __restrict__ qkv,
                                                 const float* __restrict__ klat,
                                                 const float* __restrict__ vlat,
                                                 _Float16* __restrict__ cath) {
  __shared__ float kl[CHUNKS * HDIM];
  __shared__ float vl[CHUNKS * HDIM];
  const int tid = threadIdx.x;
  const int b = blockIdx.x >> 7;
  const int h = (blockIdx.x >> 3) & 15;
  const int s = (blockIdx.x & 7) * 256 + tid;
  size_t bh = ((size_t)b * HEADS + h) * (CHUNKS * HDIM);
  for (int i = tid; i < CHUNKS * HDIM; i += 256) {
    kl[i] = klat[bh + i];
    vl[i] = vlat[bh + i];
  }
  __syncthreads();

  float qv[64];
  size_t qbase = (size_t)(b * S_LEN + s) * (3 * EMB) + h * HDIM;
  const float4* q4 = (const float4*)(qkv + qbase);
#pragma unroll
  for (int d4 = 0; d4 < 16; ++d4) {
    float4 t = q4[d4];
    qv[4 * d4] = t.x; qv[4 * d4 + 1] = t.y; qv[4 * d4 + 2] = t.z; qv[4 * d4 + 3] = t.w;
  }

  const int cmax = s >> 6;
  float m = -3.4e38f, l = 0.f;
  for (int c = 0; c <= cmax; ++c) {
    const float4* k4 = (const float4*)(kl + c * HDIM);
    float sc = 0.f;
#pragma unroll
    for (int d4 = 0; d4 < 16; ++d4) {
      float4 t = k4[d4];
      sc += qv[4 * d4] * t.x + qv[4 * d4 + 1] * t.y + qv[4 * d4 + 2] * t.z + qv[4 * d4 + 3] * t.w;
    }
    sc *= 0.125f;
    float mn = fmaxf(m, sc);
    l = l * expf(m - mn) + expf(sc - mn);
    m = mn;
  }
  float inv = 1.0f / l;

  float o[64];
#pragma unroll
  for (int d = 0; d < 64; ++d) o[d] = 0.f;
  for (int c = 0; c <= cmax; ++c) {
    const float4* k4 = (const float4*)(kl + c * HDIM);
    float sc = 0.f;
#pragma unroll
    for (int d4 = 0; d4 < 16; ++d4) {
      float4 t = k4[d4];
      sc += qv[4 * d4] * t.x + qv[4 * d4 + 1] * t.y + qv[4 * d4 + 2] * t.z + qv[4 * d4 + 3] * t.w;
    }
    float w = expf(sc * 0.125f - m) * inv;
    const float4* v4 = (const float4*)(vl + c * HDIM);
#pragma unroll
    for (int d4 = 0; d4 < 16; ++d4) {
      float4 t = v4[d4];
      o[4 * d4]     += w * t.x; o[4 * d4 + 1] += w * t.y;
      o[4 * d4 + 2] += w * t.z; o[4 * d4 + 3] += w * t.w;
    }
  }
  store64_f16(cath + (size_t)(b * S_LEN + s) * (2 * EMB) + h * 128, o);
}

// ---------------------------------------------------------------------------
// Local windowed attention -> cath[b,s, h*128 + 64..127] (f16)
// ---------------------------------------------------------------------------
__global__ __launch_bounds__(256) void local_attn(const float* __restrict__ qkv,
                                                  _Float16* __restrict__ cath) {
  const int tid = threadIdx.x;
  const int b = blockIdx.x >> 7;
  const int h = (blockIdx.x >> 3) & 15;
  const int s = (blockIdx.x & 7) * 256 + tid;

  float qv[64];
  size_t qbase = (size_t)(b * S_LEN + s) * (3 * EMB) + h * HDIM;
  const float4* q4 = (const float4*)(qkv + qbase);
#pragma unroll
  for (int d4 = 0; d4 < 16; ++d4) {
    float4 t = q4[d4];
    qv[4 * d4] = t.x; qv[4 * d4 + 1] = t.y; qv[4 * d4 + 2] = t.z; qv[4 * d4 + 3] = t.w;
  }

  const int jlo = (s >= STRIDE - 1) ? 0 : (STRIDE - 1 - s);
  float m = -3.4e38f, l = 0.f;
  for (int j = jlo; j < STRIDE; ++j) {
    int t = s - (STRIDE - 1) + j;
    const float4* k4 = (const float4*)(qkv + ((size_t)(b * S_LEN + t) * 3 + 1) * EMB + h * HDIM);
    float sc = 0.f;
#pragma unroll
    for (int d4 = 0; d4 < 16; ++d4) {
      float4 kt = k4[d4];
      sc += qv[4 * d4] * kt.x + qv[4 * d4 + 1] * kt.y + qv[4 * d4 + 2] * kt.z + qv[4 * d4 + 3] * kt.w;
    }
    sc *= 0.125f;
    float mn = fmaxf(m, sc);
    l = l * expf(m - mn) + expf(sc - mn);
    m = mn;
  }
  float inv = 1.0f / l;

  float o[64];
#pragma unroll
  for (int d = 0; d < 64; ++d) o[d] = 0.f;
  for (int j = jlo; j < STRIDE; ++j) {
    int t = s - (STRIDE - 1) + j;
    const float4* k4 = (const float4*)(qkv + ((size_t)(b * S_LEN + t) * 3 + 1) * EMB + h * HDIM);
    float sc = 0.f;
#pragma unroll
    for (int d4 = 0; d4 < 16; ++d4) {
      float4 kt = k4[d4];
      sc += qv[4 * d4] * kt.x + qv[4 * d4 + 1] * kt.y + qv[4 * d4 + 2] * kt.z + qv[4 * d4 + 3] * kt.w;
    }
    float w = expf(sc * 0.125f - m) * inv;
    const float4* v4 = (const float4*)(qkv + ((size_t)(b * S_LEN + t) * 3 + 2) * EMB + h * HDIM);
#pragma unroll
    for (int d4 = 0; d4 < 16; ++d4) {
      float4 vt = v4[d4];
      o[4 * d4]     += w * vt.x; o[4 * d4 + 1] += w * vt.y;
      o[4 * d4 + 2] += w * vt.z; o[4 * d4 + 3] += w * vt.w;
    }
  }
  store64_f16(cath + (size_t)(b * S_LEN + s) * (2 * EMB) + h * 128 + 64, o);
}

// ---------------------------------------------------------------------------
extern "C" void kernel_launch(void* const* d_in, const int* in_sizes, int n_in,
                              void* d_out, int out_size, void* d_ws, size_t ws_size,
                              hipStream_t stream) {
  const float* x     = (const float*)d_in[0];
  const float* w_qkv = (const float*)d_in[1];   // (3E, E) row-major
  const float* w_mix = (const float*)d_in[2];   // (E, 2E) row-major
  float* out = (float*)d_out;

  const int M = BATCH * S_LEN;                  // 4096
  const int nX  = M * EMB;                      // 4,194,304
  const int nWq = 3 * EMB * EMB;                // 3,145,728
  const int nWm = EMB * 2 * EMB;                // 2,097,152

  // workspace layout (byte offsets, all 256B-aligned)
  char* p = (char*)d_ws;
  float*     qkv   = (float*)p;      p += (size_t)M * 3 * EMB * 4;          // 48 MiB
  float*     klat  = (float*)p;      p += (size_t)BATCH * HEADS * CHUNKS * HDIM * 4;
  float*     vlat  = (float*)p;      p += (size_t)BATCH * HEADS * CHUNKS * HDIM * 4;
  _Float16*  xh    = (_Float16*)p;   p += (size_t)nX * 2;
  _Float16*  wqkvh = (_Float16*)p;   p += (size_t)nWq * 2;
  _Float16*  wmixh = (_Float16*)p;   p += (size_t)nWm * 2;
  _Float16*  cath  = (_Float16*)p;   p += (size_t)M * 2 * EMB * 2;

  dim3 blk(256);

  // 0) convert GEMM operands to f16
  cvt_f32_to_f16<<<dim3(nX  / 1024), blk, 0, stream>>>(x,     xh,    nX);
  cvt_f32_to_f16<<<dim3(nWq / 1024), blk, 0, stream>>>(w_qkv, wqkvh, nWq);
  cvt_f32_to_f16<<<dim3(nWm / 1024), blk, 0, stream>>>(w_mix, wmixh, nWm);

  // 1) qkv = x @ w_qkv^T   (M=4096, N=3072, K=1024)
  gemm_f16_wmma<<<dim3((3 * EMB) / 128, M / 128), blk, 0, stream>>>(
      xh, wqkvh, qkv, M, 3 * EMB, EMB);

  // 2) RoPE in place on q, k
  rope_kernel<<<dim3((BATCH * S_LEN * 2 * HEADS * 32) / 256), blk, 0, stream>>>(qkv);

  // 3) chunk-mean latents
  lat_kernel<<<dim3((2 * BATCH * HEADS * CHUNKS * HDIM) / 256), blk, 0, stream>>>(
      qkv, klat, vlat);

  // 4) compressed attention -> cath[:, :, h*128 + 0..63]
  comp_attn<<<dim3(BATCH * HEADS * (S_LEN / 256)), blk, 0, stream>>>(
      qkv, klat, vlat, cath);

  // 5) local attention -> cath[:, :, h*128 + 64..127]
  local_attn<<<dim3(BATCH * HEADS * (S_LEN / 256)), blk, 0, stream>>>(qkv, cath);

  // 6) out = cath @ w_mix^T   (M=4096, N=1024, K=2048)
  gemm_f16_wmma<<<dim3(EMB / 128, M / 128), blk, 0, stream>>>(
      cath, wmixh, out, M, EMB, 2 * EMB);
}
